// DepthwiseXCorr_1529008358039
// MI455X (gfx1250) — matrix operand order repaired and checked by reading, not compiled, not run on hardware
//
#include <hip/hip_runtime.h>
#include <hip/hip_bf16.h>

// ---------------------------------------------------------------------------
// CDNA5 (gfx1250) wave32 WMMA implementation of the SiamRPN-style depthwise
// xcorr head.  Heavy math on v_wmma_f32_16x16x32_bf16; all layout/precision
// conversion hoisted into one-time prep kernels so GEMM loops are pure
// vector loads + WMMA.  64-pixel tiles: 8 WMMAs per wave per K-step.
// ---------------------------------------------------------------------------

typedef __attribute__((ext_vector_type(16))) __bf16 v16bf;
typedef __attribute__((ext_vector_type(8)))  float  v8f;

#define EPS_BN 1e-5f

// ===========================================================================
// Prep 1: pack 3x3 conv weights [256,256,3,3] f32 into bf16 A-fragments:
//   layout [step = khw*8+ks][mtile 16][lane 32][j 16]   (72*16*32 fragments)
// A-fragment k order chosen here (contiguous, B matches): M = lane&15,
//   K = (lane>=16 ? 16 : 0) + j, within K-step ks*32.
// ===========================================================================
__global__ __launch_bounds__(256)
void pack_w3x3(const float* __restrict__ w, __bf16* __restrict__ out)
{
    const int frag = blockIdx.x * 256 + threadIdx.x;      // < 9*8*16*32 = 36864
    if (frag >= 9 * 8 * 16 * 32) return;
    const int ln    = frag & 31;
    const int mtile = (frag >> 5) & 15;
    const int ks    = (frag >> 9) & 7;
    const int khw   = frag >> 12;
    const int M     = mtile * 16 + (ln & 15);
    const int kc    = (ln >= 16) ? 16 : 0;
    v16bf v;
    #pragma unroll
    for (int j = 0; j < 16; ++j) {
        const int K = ks * 32 + kc + j;
        v[j] = (__bf16)w[((size_t)M * 256 + K) * 9 + khw];
    }
    *reinterpret_cast<v16bf*>(&out[(size_t)frag * 16]) = v;
}

// Prep 2: pack wh1 [256,256] -> [ks 8][mtile 16][lane 32][j 16]
__global__ __launch_bounds__(256)
void pack_wh1(const float* __restrict__ w, __bf16* __restrict__ out)
{
    const int frag = blockIdx.x * 256 + threadIdx.x;      // < 8*16*32 = 4096
    if (frag >= 8 * 16 * 32) return;
    const int ln    = frag & 31;
    const int mtile = (frag >> 5) & 15;
    const int ks    = frag >> 9;
    const int M     = mtile * 16 + (ln & 15);
    const int kc    = (ln >= 16) ? 16 : 0;
    v16bf v;
    #pragma unroll
    for (int j = 0; j < 16; ++j) {
        const int K = ks * 32 + kc + j;
        v[j] = (__bf16)w[(size_t)M * 256 + K];
    }
    *reinterpret_cast<v16bf*>(&out[(size_t)frag * 16]) = v;
}

// Prep 3: pack wh2 [20,256] (zero-pad M to 32) -> [ks 8][mt 2][lane 32][j 16]
__global__ __launch_bounds__(256)
void pack_wh2(const float* __restrict__ w, __bf16* __restrict__ out)
{
    const int frag = blockIdx.x * 256 + threadIdx.x;      // < 8*2*32 = 512
    if (frag >= 8 * 2 * 32) return;
    const int ln = frag & 31;
    const int mt = (frag >> 5) & 1;
    const int ks = frag >> 6;
    const int M  = mt * 16 + (ln & 15);
    const int kc = (ln >= 16) ? 16 : 0;
    v16bf v;
    #pragma unroll
    for (int j = 0; j < 16; ++j) {
        const int K = ks * 32 + kc + j;
        v[j] = (__bf16)((M < 20) ? w[(size_t)M * 256 + K] : 0.0f);
    }
    *reinterpret_cast<v16bf*>(&out[(size_t)frag * 16]) = v;
}

// Prep 4: NCHW f32 -> NHWC bf16 (channels contiguous for B-fragment loads).
__global__ __launch_bounds__(256)
void to_nhwc_bf16(const float* __restrict__ x, __bf16* __restrict__ y, int HW)
{
    const int p = blockIdx.x % HW;
    const int b = blockIdx.x / HW;
    const int c = threadIdx.x;
    y[(size_t)blockIdx.x * 256 + c] = (__bf16)x[((size_t)b * 256 + c) * HW + p];
}

// ===========================================================================
// Kernel 1: 3x3 conv (VALID) + BN + ReLU, implicit GEMM on WMMA.
//   xT : [B, H, W, 256] bf16 (NHWC)     wpk: packed A-fragments (72 steps)
//   y  : [B, 256, Ho*Wo] bf16 (NCHW, for the depthwise xcorr)
// Workgroup = 8 waves, 64-pixel tile.  Wave w owns Cout rows [32w,32w+32)
// x 64 pixels: 2 Mtiles x 4 Ntiles = 8 WMMAs per K-step.
// A fragments stream straight from global (contiguous b128 per lane).
// B tile (4 KB) double-buffered in LDS; staging uses a precomputed
// per-thread base pointer + wave-uniform per-step offset (pure SALU).
// ===========================================================================
__global__ __launch_bounds__(256)
void conv3x3_bn_relu_wmma(const __bf16* __restrict__ xT,
                          const __bf16* __restrict__ wpk,
                          const float* __restrict__ bn_g,
                          const float* __restrict__ bn_b,
                          const float* __restrict__ bn_m,
                          const float* __restrict__ bn_v,
                          __bf16* __restrict__ y,
                          int H, int W, int Ho, int Wo)
{
    const int npix        = Ho * Wo;
    const int tilesPerImg = (npix + 63) / 64;
    const int b  = blockIdx.x / tilesPerImg;
    const int pt = blockIdx.x % tilesPerImg;

    const int tid  = threadIdx.x;
    const int lane = tid & 31;
    const int wv   = tid >> 5;

    __shared__ __attribute__((aligned(32))) __bf16 ldsB[2][4 * 32 * 16];

    v8f acc[2][4] = {};

    // ---- per-thread staging base (loop-invariant) ------------------------
    const __bf16* bsrc = xT;
    bool bvalid = false;
    if (tid < 128) {
        const int nt = tid >> 5, ln = tid & 31;
        const int N  = nt * 16 + (ln & 15);
        const int P  = pt * 64 + N;
        bvalid = (P < npix);
        const int Pc = bvalid ? P : 0;
        const int py = Pc / Wo, px = Pc % Wo;
        bsrc = xT + (((size_t)b * H + py) * W + px) * 256 + ((ln >= 16) ? 16 : 0);
    }

    auto stepOff = [&](int s) -> int {      // wave-uniform scalar offset
        const int khw = s >> 3;
        return ((khw / 3) * W + (khw % 3)) * 256 + (s & 7) * 32;
    };
    auto stageB = [&](int s, int buf) {
        if (tid < 128) {
            v16bf frag;
            if (bvalid) {
                frag = *reinterpret_cast<const v16bf*>(bsrc + stepOff(s));
            } else {
                #pragma unroll
                for (int j = 0; j < 16; ++j) frag[j] = (__bf16)0.0f;
            }
            *reinterpret_cast<v16bf*>(&ldsB[buf][tid * 16]) = frag;
        }
    };

    stageB(0, 0);
    for (int s = 0; s < 72; ++s) {
        __syncthreads();
        if (s + 1 < 72) stageB(s + 1, (s + 1) & 1);
        if (s + 2 < 72 && tid < 128 && bvalid)          // gfx1250 prefetch path
            __builtin_prefetch(bsrc + stepOff(s + 2), 0, 0);
        const size_t abase = (size_t)s * (16 * 32 * 16);
        #pragma unroll
        for (int mi = 0; mi < 2; ++mi) {
            const v16bf afrag = *reinterpret_cast<const v16bf*>(
                &wpk[abase + ((size_t)((wv * 2 + mi) * 32 + lane)) * 16]);
            #pragma unroll
            for (int ni = 0; ni < 4; ++ni) {
                const v16bf bfrag = *reinterpret_cast<const v16bf*>(
                    &ldsB[s & 1][(ni * 32 + lane) * 16]);
                acc[mi][ni] = __builtin_amdgcn_wmma_f32_16x16x32_bf16(
                    false, afrag, false, bfrag,
                    (short)0, acc[mi][ni], false, false);
            }
        }
    }

    // ---- epilogue: BN + ReLU, write NCHW bf16 ----------------------------
    #pragma unroll
    for (int mi = 0; mi < 2; ++mi) {
        #pragma unroll
        for (int r = 0; r < 8; ++r) {
            const int co = (wv * 2 + mi) * 16 + r + ((lane >= 16) ? 8 : 0);
            const float scale = bn_g[co] * rsqrtf(bn_v[co] + EPS_BN);
            const float bias  = bn_b[co] - bn_m[co] * scale;
            #pragma unroll
            for (int ni = 0; ni < 4; ++ni) {
                const int N = ni * 16 + (lane & 15);
                const int P = pt * 64 + N;
                if (P < npix) {
                    float val = fmaxf(acc[mi][ni][r] * scale + bias, 0.0f);
                    y[((size_t)b * 256 + co) * npix + P] = (__bf16)val;
                }
            }
        }
    }
}

// ===========================================================================
// Kernel 2: depthwise xcorr (5x5 over 29x29 -> 25x25).  1 GFLOP -> VALU.
// One block per (b, c).  Output written pixel-major / channel-contiguous
// (fT[(b*625+p)*256 + c]) so the head's B staging is a contiguous load.
// ===========================================================================
__global__ __launch_bounds__(256)
void xcorr_dw(const __bf16* __restrict__ s,
              const __bf16* __restrict__ k,
              __bf16* __restrict__ fT)
{
    const int bc = blockIdx.x;                 // b*256 + c
    const int b  = bc >> 8;
    const int c  = bc & 255;
    const __bf16* sp = s + (size_t)bc * (29 * 29);
    const __bf16* kp = k + (size_t)bc * 25;

    __shared__ float kk[25];
    if (threadIdx.x < 25) kk[threadIdx.x] = (float)kp[threadIdx.x];
    __syncthreads();

    for (int p = threadIdx.x; p < 625; p += 256) {
        const int oy = p / 25, ox = p % 25;
        float acc = 0.0f;
        #pragma unroll
        for (int i = 0; i < 5; ++i)
            #pragma unroll
            for (int j = 0; j < 5; ++j)
                acc += (float)sp[(oy + i) * 29 + (ox + j)] * kk[i * 5 + j];
        fT[((size_t)b * 625 + p) * 256 + c] = (__bf16)acc;
    }
}

// ===========================================================================
// Kernel 3: fused head.  h = relu(bn(wh1 @ f)); out = wh2 @ h + bh2.
// GEMM1 A-fragments direct from packed global; B from fT (contiguous).
// h-tile scattered into LDS already in B-fragment layout for GEMM2.
// ===========================================================================
__global__ __launch_bounds__(256)
void head_1x1_wmma(const __bf16* __restrict__ fT,
                   const __bf16* __restrict__ wh1pk,
                   const float* __restrict__ gh,
                   const float* __restrict__ bh,
                   const float* __restrict__ mh,
                   const float* __restrict__ vh,
                   const __bf16* __restrict__ wh2pk,
                   const float* __restrict__ bh2,
                   float* __restrict__ out)
{
    constexpr int CoutF = 20, NP = 625;
    const int qt   = blockIdx.x;               // 32-pixel tile over 128*625
    const int tid  = threadIdx.x;
    const int lane = tid & 31;
    const int wv   = tid >> 5;

    __shared__ __attribute__((aligned(32))) __bf16 ldsB[2][2 * 32 * 16];
    __shared__ __attribute__((aligned(32))) __bf16 ldsH[8 * 2 * 32 * 16];

    v8f acc[2][2] = {};

    // per-thread staging base pointer (loop-invariant)
    const __bf16* bsrc = fT;
    if (tid < 64) {
        const int ln = tid & 31;
        const int nt = tid >> 5;
        const int N  = nt * 16 + (ln & 15);
        const int Q  = qt * 32 + N;               // 128*625 % 32 == 0
        bsrc = fT + (size_t)Q * 256 + ((ln >= 16) ? 16 : 0);
    }
    auto stageB = [&](int ks, int buf) {
        if (tid < 64) {
            const v16bf frag =
                *reinterpret_cast<const v16bf*>(bsrc + ks * 32);
            *reinterpret_cast<v16bf*>(&ldsB[buf][tid * 16]) = frag;
        }
    };

    stageB(0, 0);
    for (int ks = 0; ks < 8; ++ks) {
        __syncthreads();
        if (ks + 1 < 8) stageB(ks + 1, (ks + 1) & 1);
        const size_t abase = (size_t)ks * (16 * 32 * 16);
        #pragma unroll
        for (int mi = 0; mi < 2; ++mi) {
            const v16bf afrag = *reinterpret_cast<const v16bf*>(
                &wh1pk[abase + ((size_t)((wv * 2 + mi) * 32 + lane)) * 16]);
            #pragma unroll
            for (int ni = 0; ni < 2; ++ni) {
                const v16bf bfrag = *reinterpret_cast<const v16bf*>(
                    &ldsB[ks & 1][(ni * 32 + lane) * 16]);
                acc[mi][ni] = __builtin_amdgcn_wmma_f32_16x16x32_bf16(
                    false, afrag, false, bfrag,
                    (short)0, acc[mi][ni], false, false);
            }
        }
    }

    // BN + ReLU; scatter h into B-fragment layout for GEMM2
    #pragma unroll
    for (int mi = 0; mi < 2; ++mi) {
        #pragma unroll
        for (int ni = 0; ni < 2; ++ni) {
            #pragma unroll
            for (int r = 0; r < 8; ++r) {
                const int ch    = (wv * 2 + mi) * 16 + r + ((lane >= 16) ? 8 : 0);
                const float scl = gh[ch] * rsqrtf(vh[ch] + EPS_BN);
                const float bia = bh[ch] - mh[ch] * scl;
                const float hv  = fmaxf(acc[mi][ni][r] * scl + bia, 0.0f);
                const int N   = ni * 16 + (lane & 15);
                const int ks  = ch >> 5;              // GEMM2 K-step
                const int Kp  = ch & 31;
                const int ln2 = ((Kp >= 16) ? 16 : 0) + (N & 15);
                const int j2  = Kp & 15;
                ldsH[((ks * 2 + ni) * 32 + ln2) * 16 + j2] = (__bf16)hv;
            }
        }
    }
    __syncthreads();

    // GEMM2: waves 0..3 each own one (Mtile, Ntile); A2 direct from global
    if (wv < 4) {
        const int mt2 = wv >> 1, nt2 = wv & 1;
        v8f acc2 = {};
        #pragma unroll
        for (int ks = 0; ks < 8; ++ks) {
            const v16bf a2 = *reinterpret_cast<const v16bf*>(
                &wh2pk[((size_t)((ks * 2 + mt2) * 32 + lane)) * 16]);
            const v16bf b2 = *reinterpret_cast<const v16bf*>(
                &ldsH[((ks * 2 + nt2) * 32 + lane) * 16]);
            acc2 = __builtin_amdgcn_wmma_f32_16x16x32_bf16(
                false, a2, false, b2, (short)0, acc2, false, false);
        }
        #pragma unroll
        for (int r = 0; r < 8; ++r) {
            const int co = mt2 * 16 + r + ((lane >= 16) ? 8 : 0);
            if (co < CoutF) {
                const int N  = nt2 * 16 + (lane & 15);
                const int Q  = qt * 32 + N;
                const int bb = Q / NP, p = Q % NP;
                out[((size_t)bb * CoutF + co) * NP + p] = acc2[r] + bh2[co];
            }
        }
    }
}

// ===========================================================================
// Host launcher
// ===========================================================================
extern "C" void kernel_launch(void* const* d_in, const int* in_sizes, int n_in,
                              void* d_out, int out_size, void* d_ws, size_t ws_size,
                              hipStream_t stream) {
    const float* kernel_in = (const float*)d_in[0];
    const float* search_in = (const float*)d_in[1];
    const float* wk = (const float*)d_in[2];
    const float* gk = (const float*)d_in[3];
    const float* bk = (const float*)d_in[4];
    const float* mk = (const float*)d_in[5];
    const float* vk = (const float*)d_in[6];
    const float* wsw = (const float*)d_in[7];
    const float* gs = (const float*)d_in[8];
    const float* bs = (const float*)d_in[9];
    const float* ms = (const float*)d_in[10];
    const float* vs = (const float*)d_in[11];
    const float* wh1 = (const float*)d_in[12];
    const float* gh  = (const float*)d_in[13];
    const float* bh  = (const float*)d_in[14];
    const float* mh  = (const float*)d_in[15];
    const float* vh  = (const float*)d_in[16];
    const float* wh2 = (const float*)d_in[17];
    const float* bh2 = (const float*)d_in[18];
    float* out = (float*)d_out;

    // ---- workspace layout (bytes, all offsets 32B-aligned) ---------------
    char* ws = (char*)d_ws;
    size_t off = 0;
    auto alloc = [&](size_t bytes) {
        char* p = ws + off;
        off += (bytes + 255) & ~(size_t)255;
        return p;
    };
    __bf16* kbf   = (__bf16*)alloc((size_t)128 * 256 * 25 * 2);        //  1.6 MB
    __bf16* sbf   = (__bf16*)alloc((size_t)128 * 256 * 841 * 2);       // 55.1 MB
    __bf16* fTb   = (__bf16*)alloc((size_t)128 * 625 * 256 * 2);       // 41.0 MB
    __bf16* xTk   = (__bf16*)alloc((size_t)128 * 49 * 256 * 2);        //  3.2 MB
    __bf16* xTs   = (__bf16*)alloc((size_t)128 * 961 * 256 * 2);       // 63.0 MB
    __bf16* wpkK  = (__bf16*)alloc((size_t)36864 * 16 * 2);            //  1.2 MB
    __bf16* wpkS  = (__bf16*)alloc((size_t)36864 * 16 * 2);            //  1.2 MB
    __bf16* wh1pk = (__bf16*)alloc((size_t)4096 * 16 * 2);             //  128 KB
    __bf16* wh2pk = (__bf16*)alloc((size_t)512 * 16 * 2);              //   16 KB

    // ---- one-time packing / layout transforms ----------------------------
    pack_w3x3<<<144, 256, 0, stream>>>(wk,  wpkK);
    pack_w3x3<<<144, 256, 0, stream>>>(wsw, wpkS);
    pack_wh1 <<< 16, 256, 0, stream>>>(wh1, wh1pk);
    pack_wh2 <<<  2, 256, 0, stream>>>(wh2, wh2pk);
    to_nhwc_bf16<<<128 * 49,  256, 0, stream>>>(kernel_in, xTk, 49);
    to_nhwc_bf16<<<128 * 961, 256, 0, stream>>>(search_in, xTs, 961);

    // ---- pipeline --------------------------------------------------------
    // 1) kernel branch: [128,256,7,7] -> [128,256,5,5]  (25 pix -> 1 tile)
    conv3x3_bn_relu_wmma<<<128 * 1, 256, 0, stream>>>(
        xTk, wpkK, gk, bk, mk, vk, kbf, 7, 7, 5, 5);
    // 2) search branch: [128,256,31,31] -> [128,256,29,29] (14 tiles/img)
    conv3x3_bn_relu_wmma<<<128 * 14, 256, 0, stream>>>(
        xTs, wpkS, gs, bs, ms, vs, sbf, 31, 31, 29, 29);
    // 3) depthwise xcorr -> fT [b][p][c] bf16
    xcorr_dw<<<128 * 256, 256, 0, stream>>>(sbf, kbf, fTb);
    // 4) fused 1x1 head: 80000 pixels / 32 = 2500 blocks
    head_1x1_wmma<<<2500, 256, 0, stream>>>(
        fTb, wh1pk, gh, bh, mh, vh, wh2pk, bh2, out);
}
